// Model_91070486545135
// MI455X (gfx1250) — compile-verified, matrix operand order checked by reading, not run on hardware
//
#include <hip/hip_runtime.h>

#define B_  32
#define T_  256
#define DIN 512
#define H_  1024
#define V_  10000

typedef __attribute__((ext_vector_type(16))) __bf16 bf16x16;
typedef __attribute__((ext_vector_type(8)))  __bf16 bf16x8;
typedef __attribute__((ext_vector_type(8)))  float  f32x8;

__device__ __forceinline__ __bf16 f2bf(float f) {
  unsigned u = __builtin_bit_cast(unsigned, f);
  unsigned r = u + 0x7fffu + ((u >> 16) & 1u);   // round-to-nearest-even
  unsigned short h = (unsigned short)(r >> 16);
  return __builtin_bit_cast(__bf16, h);
}

// fast sigmoid / tanh (v_exp + v_rcp, no IEEE division chain)
__device__ __forceinline__ float sigm(float x) {
  return __builtin_amdgcn_rcpf(1.f + __expf(-x));
}
__device__ __forceinline__ float tanh_f(float x) {
  return 2.f * sigm(2.f * x) - 1.f;
}

// 16x32 bf16 fragment, row-major source; row = lane&15, K-halves per ISA layout.
__device__ __forceinline__ void load_frag(const __bf16* base, int ld, int k0,
                                          int lane, bf16x16& f) {
  const __bf16* p = base + (size_t)(lane & 15) * ld + k0 + ((lane >> 4) << 3);
  bf16x8 lo = *(const bf16x8*)(p);        // 16B load
  bf16x8 hi = *(const bf16x8*)(p + 16);   // 16B load
#pragma unroll
  for (int i = 0; i < 8; ++i) { f[i] = lo[i]; f[8 + i] = hi[i]; }
}

// Y[32,16-tile] = A[32,K] * W^T ; weight fragment loaded ONCE per K-step,
// reused for both M-tiles (rows 0-15 and 16-31).
__device__ __forceinline__ void gemm2(const __bf16* A, int lda, const __bf16* W,
                                      int K, int nbase, int lane,
                                      f32x8& c0, f32x8& c1) {
  const __bf16* a1 = A + (size_t)16 * lda;
  const __bf16* bb = W + (size_t)nbase * K;
  for (int k0 = 0; k0 < K; k0 += 32) {
    bf16x16 b, x0, x1;
    load_frag(bb, K, k0, lane, b);
    load_frag(A,  lda, k0, lane, x0);
    load_frag(a1, lda, k0, lane, x1);
    c0 = __builtin_amdgcn_wmma_f32_16x16x32_bf16(false, x0, false, b, (short)0, c0,
                                                 false, false);
    c1 = __builtin_amdgcn_wmma_f32_16x16x32_bf16(false, x1, false, b, (short)0, c1,
                                                 false, false);
  }
}

// ---------------- kernels ----------------

__global__ void k_convert(const float* __restrict__ src, __bf16* __restrict__ dst, int n) {
  int i = blockIdx.x * blockDim.x + threadIdx.x;
  if (i < n) dst[i] = f2bf(src[i]);
}

__global__ void k_zero32(float* p, int n) {
  int i = blockIdx.x * blockDim.x + threadIdx.x;
  if (i < n) p[i] = 0.f;
}

__global__ void k_zero16(unsigned short* p, int n) {
  int i = blockIdx.x * blockDim.x + threadIdx.x;
  if (i < n) p[i] = 0;
}

__global__ void k_embed(const int* __restrict__ ids, const float* __restrict__ emb,
                        float* __restrict__ e, __bf16* __restrict__ ebf) {
  int i = blockIdx.x * blockDim.x + threadIdx.x;   // over B*T*DIN
  if (i >= B_ * T_ * DIN) return;
  int d  = i & (DIN - 1);
  int bt = i >> 9;                                  // DIN = 512
  float v = emb[(size_t)ids[bt] * DIN + d];
  e[i] = v;
  ebf[i] = f2bf(v);
}

// mogrifier: tout = 2*sigmoid(A*W^T + bias) * tin ; also writes bf16 shadow
__global__ void k_mog(const __bf16* __restrict__ A, int lda,
                      const __bf16* __restrict__ W, int K,
                      const float* __restrict__ bias,
                      const float* __restrict__ tin, float* __restrict__ tout,
                      __bf16* __restrict__ tbf, int ldt) {
  int lane = threadIdx.x;
  int nbase = blockIdx.x << 4;
  f32x8 c0 = {}, c1 = {};
  gemm2(A, lda, W, K, nbase, lane, c0, c1);
  int n  = nbase + (lane & 15);
  int mb = (lane >> 4) << 3;
  float bv = bias[n];
#pragma unroll
  for (int r = 0; r < 8; ++r) {
    size_t i0 = (size_t)(mb + r) * ldt + n;
    size_t i1 = (size_t)(16 + mb + r) * ldt + n;
    float v0 = 2.f * sigm(c0[r] + bv) * tin[i0];
    float v1 = 2.f * sigm(c1[r] + bv) * tin[i1];
    tout[i0] = v0; tbf[i0] = f2bf(v0);
    tout[i1] = v1; tbf[i1] = f2bf(v1);
  }
}

// gates = X*Wih^T + H*Whh^T + bih + bhh   (N = 4H)
__global__ void k_gates(const __bf16* __restrict__ X, int ldx, int Kx,
                        const __bf16* __restrict__ Wih,
                        const __bf16* __restrict__ Hs, int ldh, int Kh,
                        const __bf16* __restrict__ Whh,
                        const float* __restrict__ bih, const float* __restrict__ bhh,
                        float* __restrict__ gates) {
  int lane = threadIdx.x;
  int nbase = blockIdx.x << 4;
  f32x8 c0 = {}, c1 = {};
  gemm2(X,  ldx, Wih, Kx, nbase, lane, c0, c1);
  gemm2(Hs, ldh, Whh, Kh, nbase, lane, c0, c1);
  int n  = nbase + (lane & 15);
  int mb = (lane >> 4) << 3;
  float bv = bih[n] + bhh[n];
#pragma unroll
  for (int r = 0; r < 8; ++r) {
    gates[(size_t)(mb + r) * (4 * H_) + n]      = c0[r] + bv;
    gates[(size_t)(16 + mb + r) * (4 * H_) + n] = c1[r] + bv;
  }
}

// LSTM elementwise: c,h (+bf16 shadow) in place; optional hidden_states stream
__global__ void k_lstm(const float* __restrict__ gates, float* __restrict__ c,
                       float* __restrict__ h, __bf16* __restrict__ hbf,
                       float* __restrict__ hs, int ldhs) {
  int i = blockIdx.x * blockDim.x + threadIdx.x;   // over B*H
  if (i >= B_ * H_) return;
  int b = i >> 10, j = i & (H_ - 1);
  const float* g = gates + (size_t)b * 4 * H_;
  float ig = sigm(g[j]);
  float fg = sigm(g[H_ + j]);
  float gg = tanh_f(g[2 * H_ + j]);
  float og = sigm(g[3 * H_ + j]);
  float cn = fg * c[i] + ig * gg;
  float hn = og * tanh_f(cn);
  c[i] = cn;
  h[i] = hn;
  hbf[i] = f2bf(hn);
  if (hs) hs[(size_t)b * ldhs + j] = hn;
}

// FC head: out = A*W^T + bias
__global__ void k_fc(const __bf16* __restrict__ A, int lda,
                     const __bf16* __restrict__ W, int K,
                     const float* __restrict__ bias, float* __restrict__ out, int ldo) {
  int lane = threadIdx.x;
  int nbase = blockIdx.x << 4;
  f32x8 c0 = {}, c1 = {};
  gemm2(A, lda, W, K, nbase, lane, c0, c1);
  int n  = nbase + (lane & 15);
  int mb = (lane >> 4) << 3;
  float bv = bias[n];
#pragma unroll
  for (int r = 0; r < 8; ++r) {
    out[(size_t)(mb + r) * ldo + n]      = c0[r] + bv;
    out[(size_t)(16 + mb + r) * ldo + n] = c1[r] + bv;
  }
}

// ---------------- launcher ----------------

static inline __bf16* cvt(const float* src, char* ws, size_t& off, int n, hipStream_t s) {
  __bf16* dst = (__bf16*)(ws + off);
  off += (size_t)n * 2;
  off = (off + 255) & ~(size_t)255;
  k_convert<<<(n + 255) / 256, 256, 0, s>>>(src, dst, n);
  return dst;
}

extern "C" void kernel_launch(void* const* d_in, const int* in_sizes, int n_in,
                              void* d_out, int out_size, void* d_ws, size_t ws_size,
                              hipStream_t stream) {
  const int*   ids    = (const int*)d_in[0];
  const float* emb    = (const float*)d_in[1];
  const float* m1_xw  = (const float*)d_in[2];
  const float* m1_xb  = (const float*)d_in[3];
  const float* m1_hw  = (const float*)d_in[4];
  const float* m1_hb  = (const float*)d_in[5];
  const float* l1_wih = (const float*)d_in[6];
  const float* l1_whh = (const float*)d_in[7];
  const float* l1_bih = (const float*)d_in[8];
  const float* l1_bhh = (const float*)d_in[9];
  const float* m2_xw  = (const float*)d_in[10];
  const float* m2_xb  = (const float*)d_in[11];
  const float* m2_hw  = (const float*)d_in[12];
  const float* m2_hb  = (const float*)d_in[13];
  const float* l2_wih = (const float*)d_in[14];
  const float* l2_whh = (const float*)d_in[15];
  const float* l2_bih = (const float*)d_in[16];
  const float* l2_bhh = (const float*)d_in[17];
  const float* fc_w   = (const float*)d_in[18];
  const float* fc_b   = (const float*)d_in[19];

  float* outputs = (float*)d_out;                         // [B,T,V]
  float* hidden  = outputs + (size_t)B_ * T_ * V_;        // [B,T,H]

  char* ws = (char*)d_ws;
  size_t off = 0;

  // one-time bf16 weight copies
  __bf16* b_m1xw = cvt(m1_xw, ws, off, 3 * DIN * H_, stream);
  __bf16* b_m1hw = cvt(m1_hw, ws, off, 2 * H_ * DIN, stream);
  __bf16* b_l1ih = cvt(l1_wih, ws, off, 4 * H_ * DIN, stream);
  __bf16* b_l1hh = cvt(l1_whh, ws, off, 4 * H_ * H_, stream);
  __bf16* b_m2xw = cvt(m2_xw, ws, off, 3 * H_ * H_, stream);
  __bf16* b_m2hw = cvt(m2_hw, ws, off, 2 * H_ * H_, stream);
  __bf16* b_l2ih = cvt(l2_wih, ws, off, 4 * H_ * H_, stream);
  __bf16* b_l2hh = cvt(l2_whh, ws, off, 4 * H_ * H_, stream);
  __bf16* b_fcw  = cvt(fc_w,  ws, off, V_ * H_, stream);

  // fp32 masters
  float* e     = (float*)(ws + off); off += (size_t)B_ * T_ * DIN * 4;   // [B,T,DIN]
  float* h1    = (float*)(ws + off); off += B_ * H_ * 4;
  float* c1    = (float*)(ws + off); off += B_ * H_ * 4;
  float* h2    = (float*)(ws + off); off += B_ * H_ * 4;
  float* c2    = (float*)(ws + off); off += B_ * H_ * 4;
  float* xbuf  = (float*)(ws + off); off += B_ * H_ * 4;
  float* gates = (float*)(ws + off); off += (size_t)B_ * 4 * H_ * 4;
  // bf16 activation shadows (matmul operands)
  __bf16* ebf  = (__bf16*)(ws + off); off += (size_t)B_ * T_ * DIN * 2;
  __bf16* h1b  = (__bf16*)(ws + off); off += B_ * H_ * 2;
  __bf16* h2b  = (__bf16*)(ws + off); off += B_ * H_ * 2;
  __bf16* xbfb = (__bf16*)(ws + off); off += B_ * H_ * 2;

  {
    int nz = 5 * B_ * H_ + B_ * 4 * H_;   // h1..xbuf + gates (contiguous fp32)
    k_zero32<<<(nz + 255) / 256, 256, 0, stream>>>(h1, nz);
    int nb = 3 * B_ * H_;                 // h1b,h2b,xbfb (contiguous bf16)
    k_zero16<<<(nb + 255) / 256, 256, 0, stream>>>((unsigned short*)h1b, nb);
  }
  {
    int n = B_ * T_ * DIN;
    k_embed<<<(n + 255) / 256, 256, 0, stream>>>(ids, emb, e, ebf);
  }

  const int ldE  = T_ * DIN;   // row stride of e / ebf slice (fixed t)
  const int ldHS = T_ * H_;
  const int ldO  = T_ * V_;

  dim3 blk(32, 1, 1);
  dim3 gX(DIN / 16);        // 32 tiles
  dim3 gH(H_ / 16);         // 64
  dim3 gG(4 * H_ / 16);     // 256
  dim3 gV(V_ / 16);         // 625

  for (int t = 0; t < T_; ++t) {
    float*  xs  = e   + (size_t)t * DIN;   // fp32 master slice, ld = ldE
    __bf16* xsb = ebf + (size_t)t * DIN;   // bf16 shadow slice, ld = ldE

    // ---- mogrifier 1 (x<->h1), LSTM 1 ----
    k_mog<<<gX, blk, 0, stream>>>(h1b, H_, b_m1xw + 0 * (size_t)DIN * H_, H_,
                                  m1_xb + 0 * DIN, xs, xs, xsb, ldE);
    k_mog<<<gH, blk, 0, stream>>>(xsb, ldE, b_m1hw + 0 * (size_t)H_ * DIN, DIN,
                                  m1_hb + 0 * H_, h1, h1, h1b, H_);
    k_mog<<<gX, blk, 0, stream>>>(h1b, H_, b_m1xw + 1 * (size_t)DIN * H_, H_,
                                  m1_xb + 1 * DIN, xs, xs, xsb, ldE);
    k_mog<<<gH, blk, 0, stream>>>(xsb, ldE, b_m1hw + 1 * (size_t)H_ * DIN, DIN,
                                  m1_hb + 1 * H_, h1, h1, h1b, H_);
    k_mog<<<gX, blk, 0, stream>>>(h1b, H_, b_m1xw + 2 * (size_t)DIN * H_, H_,
                                  m1_xb + 2 * DIN, xs, xs, xsb, ldE);

    k_gates<<<gG, blk, 0, stream>>>(xsb, ldE, DIN, b_l1ih, h1b, H_, H_, b_l1hh,
                                    l1_bih, l1_bhh, gates);
    k_lstm<<<(B_ * H_ + 255) / 256, 256, 0, stream>>>(gates, c1, h1, h1b,
                                                      (float*)nullptr, 0);

    // ---- mogrifier 2 (x=h1n <-> h2), LSTM 2 ----
    k_mog<<<gH, blk, 0, stream>>>(h2b, H_, b_m2xw + 0 * (size_t)H_ * H_, H_,
                                  m2_xb + 0 * H_, h1, xbuf, xbfb, H_);  // copy-on-write
    k_mog<<<gH, blk, 0, stream>>>(xbfb, H_, b_m2hw + 0 * (size_t)H_ * H_, H_,
                                  m2_hb + 0 * H_, h2, h2, h2b, H_);
    k_mog<<<gH, blk, 0, stream>>>(h2b, H_, b_m2xw + 1 * (size_t)H_ * H_, H_,
                                  m2_xb + 1 * H_, xbuf, xbuf, xbfb, H_);
    k_mog<<<gH, blk, 0, stream>>>(xbfb, H_, b_m2hw + 1 * (size_t)H_ * H_, H_,
                                  m2_hb + 1 * H_, h2, h2, h2b, H_);
    k_mog<<<gH, blk, 0, stream>>>(h2b, H_, b_m2xw + 2 * (size_t)H_ * H_, H_,
                                  m2_xb + 2 * H_, xbuf, xbuf, xbfb, H_);

    k_gates<<<gG, blk, 0, stream>>>(xbfb, H_, H_, b_l2ih, h2b, H_, H_, b_l2hh,
                                    l2_bih, l2_bhh, gates);
    k_lstm<<<(B_ * H_ + 255) / 256, 256, 0, stream>>>(gates, c2, h2, h2b,
                                                      hidden + (size_t)t * H_, ldHS);

    k_fc<<<gV, blk, 0, stream>>>(h2b, H_, b_fcw, H_, fc_b,
                                 outputs + (size_t)t * V_, ldO);
  }
}
